// DecoderBlock_26190710571416
// MI455X (gfx1250) — matrix operand order repaired and checked by reading, not compiled, not run on hardware
//
#include <hip/hip_runtime.h>
#include <hip/hip_bf16.h>

typedef __attribute__((ext_vector_type(16))) _Float16 v16h;
typedef __attribute__((ext_vector_type(8)))  float    v8f;

constexpr int  Bz = 8, Tz = 512, Ez = 1024, Hz = 16, Az = 64, NFz = 64, HSz = 64, MLPz = 4096;
constexpr long BTE = (long)Bz * Tz * Ez;   // 4,194,304

__device__ __forceinline__ float geluf(float x) {
    return 0.5f * x * (1.0f + erff(x * 0.70710678118654752f));
}

// ---------------------------------------------------------------------------
// Generic tiled WMMA GEMM: C = epilogue(A[MxK] @ B[KxN] + bias)
// EPI: 0 = bias, 1 = bias+GELU, 2 = resid + alpha*(acc+bias)
// PRECONDITION (all launches satisfy it): M%64==0, N%64==0, K%32==0,
// A/B 16-byte aligned rows (K,N multiples of 4).
// 64x64 tile, BK=32, 4 waves x (2x2) wmma 16x16x32 f16, LDS double-buffered.
// ---------------------------------------------------------------------------
template <int EPI>
__global__ __launch_bounds__(128)
void k_gemm(const float* __restrict__ A, const float* __restrict__ Bw,
            const float* __restrict__ bias, int biasRow,
            const float* __restrict__ resid, const float* __restrict__ alpha,
            int alphaStride, int rowsPerB,
            float* __restrict__ C, int M, int N, int K,
            long sA, long sB, long sC)
{
    __shared__ _Float16 As[2][64][40];   // [m][k]
    __shared__ _Float16 Bs[2][64][40];   // [n][k]   (transposed B tile)

    const int z = blockIdx.z;
    A  += (long)z * sA;
    Bw += (long)z * sB;
    C  += (long)z * sC;
    const float* residp = (EPI == 2) ? (resid + (long)z * sC) : nullptr;

    const int tid  = threadIdx.x;
    const int wave = tid >> 5, lane = tid & 31;
    const int wm   = (wave >> 1) * 32, wn = (wave & 1) * 32;
    const int half = lane >> 4, l16 = lane & 15;
    const int bM   = blockIdx.y * 64, bN = blockIdx.x * 64;

    // tile-load coordinates: A = 64x32 (8 float4/row), B = 32x64 (16 float4/row)
    const int ar = tid >> 3, ac = (tid & 7) * 4;    // A rows ar+{0,16,32,48}
    const int br = tid >> 4, bc = (tid & 15) * 4;   // B k-rows br+{0,8,16,24}
    const int akk = (ac & 7) | ((ac & 8) << 1) | ((ac & 16) >> 1); // 4 contiguous halves

    float4 ra[4], rb[4];

    auto loadT = [&](int k0) {
#pragma unroll
        for (int i = 0; i < 4; ++i)
            ra[i] = *(const float4*)(A + (long)(bM + ar + i * 16) * K + k0 + ac);
#pragma unroll
        for (int i = 0; i < 4; ++i)
            rb[i] = *(const float4*)(Bw + (long)(k0 + br + i * 8) * N + bN + bc);
    };
    auto storeT = [&](int buf) {
#pragma unroll
        for (int i = 0; i < 4; ++i) {
            _Float16* p = &As[buf][ar + i * 16][ac];
            p[0] = (_Float16)ra[i].x; p[1] = (_Float16)ra[i].y;
            p[2] = (_Float16)ra[i].z; p[3] = (_Float16)ra[i].w;
        }
#pragma unroll
        for (int i = 0; i < 4; ++i) {
            int r = br + i * 8;
            Bs[buf][bc + 0][r] = (_Float16)rb[i].x;
            Bs[buf][bc + 1][r] = (_Float16)rb[i].y;
            Bs[buf][bc + 2][r] = (_Float16)rb[i].z;
            Bs[buf][bc + 3][r] = (_Float16)rb[i].w;
        }
    };

    v8f acc[2][2] = {};

    loadT(0);
    storeT(0);
    __syncthreads();

    for (int k0 = 0; k0 < K; k0 += 32) {
        const int  cur  = (k0 >> 5) & 1;
        const bool more = (k0 + 32) < K;
        if (more) loadT(k0 + 32);       // global loads in flight during WMMA

        v16h af[2], bf[2];
#pragma unroll
        for (int i = 0; i < 2; ++i) {
            const _Float16* p = &As[cur][wm + i * 16 + l16][half * 16];
#pragma unroll
            for (int h = 0; h < 16; ++h) af[i][h] = p[h];
        }
#pragma unroll
        for (int j = 0; j < 2; ++j) {
            const _Float16* p = &Bs[cur][wn + j * 16 + l16][half * 16];
#pragma unroll
            for (int h = 0; h < 16; ++h) bf[j][h] = p[h];
        }
#pragma unroll
        for (int i = 0; i < 2; ++i)
#pragma unroll
            for (int j = 0; j < 2; ++j)
                acc[i][j] = __builtin_amdgcn_wmma_f32_16x16x32_f16(
                    false, af[i], false, bf[j], (short)0, acc[i][j], false, false);

        if (more) storeT(cur ^ 1);
        __syncthreads();
    }

    // epilogue + store (C/D layout: VGPR v -> row v + 8*half, col = l16)
#pragma unroll
    for (int i = 0; i < 2; ++i) {
#pragma unroll
        for (int j = 0; j < 2; ++j) {
            const int gcol = bN + wn + j * 16 + l16;
#pragma unroll
            for (int v = 0; v < 8; ++v) {
                const int grow = bM + wm + i * 16 + half * 8 + v;
                float val = acc[i][j][v];
                val += biasRow ? bias[grow] : bias[gcol];
                if (EPI == 1) val = geluf(val);
                if (EPI == 2) {
                    float al = alpha[(grow / rowsPerB) * alphaStride + gcol];
                    val = residp[(long)grow * N + gcol] + al * val;
                }
                C[(long)grow * N + gcol] = val;
            }
        }
    }
}

// ---------------------------------------------------------------------------
// timestep -> silu(sinusoidal embedding)
// ---------------------------------------------------------------------------
__global__ void k_temb(const int* __restrict__ ts, float* __restrict__ silu) {
    int b = blockIdx.x, i = threadIdx.x;   // 512 threads
    float freq = __expf(-logf(10000.0f) * (float)i / 511.0f);
    float a = (float)ts[b] * freq;
    float s = sinf(a), c = cosf(a);
    silu[b * Ez + i]       = s / (1.0f + __expf(-s));
    silu[b * Ez + 512 + i] = c / (1.0f + __expf(-c));
}

// small scalar matmul: out[MxN] = A[MxK] @ W[KxN] + bias
__global__ void k_smallmm(const float* __restrict__ A, const float* __restrict__ W,
                          const float* __restrict__ bias, float* __restrict__ out,
                          int M, int K, int N) {
    int idx = blockIdx.x * blockDim.x + threadIdx.x;
    if (idx >= M * N) return;
    int m = idx / N, n = idx % N;
    float acc = bias[n];
    for (int k = 0; k < K; ++k) acc += A[(long)m * K + k] * W[(long)k * N + n];
    out[idx] = acc;
}

// adaLN normalize: out = ((x-mu)*rsqrt(var+eps))*(1+scale) + shift
__global__ __launch_bounds__(256)
void k_adaln(const float* __restrict__ x, const float* __restrict__ h, float* __restrict__ out) {
    __shared__ float r1[256], r2[256];
    int row = blockIdx.x;                // b*T + t
    int b = row / Tz;
    const float* xr = x + (long)row * Ez;
    int tid = threadIdx.x;
    float v[4]; float s = 0.f, ss = 0.f;
#pragma unroll
    for (int i = 0; i < 4; ++i) { float t = xr[tid + i * 256]; v[i] = t; s += t; ss += t * t; }
    r1[tid] = s; r2[tid] = ss; __syncthreads();
    for (int st = 128; st > 0; st >>= 1) {
        if (tid < st) { r1[tid] += r1[tid + st]; r2[tid] += r2[tid + st]; }
        __syncthreads();
    }
    float mu  = r1[0] * (1.0f / Ez);
    float var = r2[0] * (1.0f / Ez) - mu * mu;
    float rs  = rsqrtf(var + 1e-5f);
    const float* shift = h + (long)b * 3 * Ez;
    const float* scale = shift + Ez;
    float* orow = out + (long)row * Ez;
#pragma unroll
    for (int i = 0; i < 4; ++i) {
        int e = tid + i * 256;
        orow[e] = (v[i] - mu) * rs * (1.0f + scale[e]) + shift[e];
    }
}

// qa[h,a,:] = agents[h,a,:] @ qw + qb
__global__ __launch_bounds__(64)
void k_qa(const float* __restrict__ agents, const float* __restrict__ W,
          const float* __restrict__ bias, float* __restrict__ qa) {
    __shared__ float r[64];
    int h = blockIdx.x >> 6, a = blockIdx.x & 63, sp = threadIdx.x;
    r[sp] = agents[((h * Az + a) * HSz) + sp];
    __syncthreads();
    float acc = bias[sp];
#pragma unroll 8
    for (int s = 0; s < 64; ++s) acc += r[s] * W[s * HSz + sp];
    qa[(h * Az + a) * HSz + sp] = acc;
}

// per-(b,h,t) 64x64 head projection: out[(b,h,t),:] = row @ W + bias (head-major out)
__global__ __launch_bounds__(64)
void k_headproj(const float* __restrict__ base, int rowStride, int colOff,
                const float* __restrict__ W, const float* __restrict__ bias,
                float* __restrict__ out) {
    __shared__ float r[64];
    int bid = blockIdx.x;                 // b*H*T + h*T + t
    int t = bid % Tz, h = (bid / Tz) % Hz, b = bid / (Tz * Hz);
    int sp = threadIdx.x;
    const float* row = base + (long)(b * Tz + t) * rowStride + colOff + h * HSz;
    r[sp] = row[sp];
    __syncthreads();
    float acc = bias[sp];
#pragma unroll 8
    for (int s = 0; s < 64; ++s) acc += r[s] * W[s * HSz + sp];
    out[(long)bid * HSz + sp] = acc;
}

// s1[b,h,a,:] = softmax_t( qa[h,a,:] . ka[b,h,t,:] * 1/8 )
__global__ __launch_bounds__(256)
void k_s1(const float* __restrict__ qa, const float* __restrict__ ka, float* __restrict__ s1) {
    __shared__ float q[64]; __shared__ float red[256];
    int bid = blockIdx.x;                 // (b*H + h)*A + a
    int a = bid & 63, h = (bid >> 6) & 15, bh = bid >> 6;
    int tid = threadIdx.x;
    if (tid < 64) q[tid] = qa[(h * Az + a) * HSz + tid];
    __syncthreads();
    const float* kb = ka + (long)bh * Tz * HSz;
    float sc[2];
#pragma unroll
    for (int i = 0; i < 2; ++i) {
        int t = tid + i * 256;
        const float* kr = kb + (long)t * HSz;
        float acc = 0.f;
#pragma unroll 8
        for (int s = 0; s < 64; ++s) acc += q[s] * kr[s];
        sc[i] = acc * 0.125f;
    }
    float mx = fmaxf(sc[0], sc[1]);
    red[tid] = mx; __syncthreads();
    for (int st = 128; st > 0; st >>= 1) {
        if (tid < st) red[tid] = fmaxf(red[tid], red[tid + st]);
        __syncthreads();
    }
    mx = red[0]; __syncthreads();
    float e0 = __expf(sc[0] - mx), e1 = __expf(sc[1] - mx);
    red[tid] = e0 + e1; __syncthreads();
    for (int st = 128; st > 0; st >>= 1) {
        if (tid < st) red[tid] += red[tid + st];
        __syncthreads();
    }
    float inv = 1.0f / red[0];
    float* srow = s1 + (long)bid * Tz;
    srow[tid] = e0 * inv;
    srow[tid + 256] = e1 * inv;
}

// c1[b,h,a,:] = s1[b,h,a,:] @ va[b,h,:,:]
__global__ __launch_bounds__(64)
void k_c1(const float* __restrict__ s1, const float* __restrict__ va, float* __restrict__ c1) {
    int bid = blockIdx.x, sp = threadIdx.x;
    int bh = bid >> 6;
    const float* srow = s1 + (long)bid * Tz;
    const float* vb = va + (long)bh * Tz * HSz;
    float acc = 0.f;
    for (int t = 0; t < Tz; ++t) acc += srow[t] * vb[(long)t * HSz + sp];
    c1[(long)bid * HSz + sp] = acc;
}

// fused s2 = softmax_a(q.c1^T/8), y = s2 @ c1 (merged heads)
__global__ __launch_bounds__(64)
void k_attn2(const float* __restrict__ qbase, int rowStride, int colOff,
             const float* __restrict__ c1, float* __restrict__ y) {
    __shared__ float qr[64]; __shared__ float c1s[64][65];
    __shared__ float red[64]; __shared__ float s2[64];
    int bid = blockIdx.x;                 // b*H*T + h*T + t
    int t = bid % Tz, h = (bid / Tz) % Hz, b = bid / (Tz * Hz);
    int sp = threadIdx.x;
    const float* q = qbase + (long)(b * Tz + t) * rowStride + colOff + h * HSz;
    qr[sp] = q[sp];
    const float* cb = c1 + (long)(b * Hz + h) * Az * HSz;
    for (int a = 0; a < 64; ++a) c1s[a][sp] = cb[a * HSz + sp];
    __syncthreads();
    float acc = 0.f;
#pragma unroll 8
    for (int s = 0; s < 64; ++s) acc += qr[s] * c1s[sp][s];
    acc *= 0.125f;
    red[sp] = acc; __syncthreads();
    for (int st = 32; st > 0; st >>= 1) {
        if (sp < st) red[sp] = fmaxf(red[sp], red[sp + st]);
        __syncthreads();
    }
    float mx = red[0]; __syncthreads();
    float e = __expf(acc - mx);
    red[sp] = e; __syncthreads();
    for (int st = 32; st > 0; st >>= 1) {
        if (sp < st) red[sp] += red[sp + st];
        __syncthreads();
    }
    s2[sp] = e / red[0]; __syncthreads();
    float o = 0.f;
#pragma unroll 8
    for (int a = 0; a < 64; ++a) o += s2[a] * c1s[a][sp];
    y[(long)(b * Tz + t) * Ez + h * HSz + sp] = o;
}

// conv1 over x1 = p[:, :T, :]  (NCH, C=T, kernel 3, pad 1) + GELU
__global__ __launch_bounds__(256)
void k_conv1(const float* __restrict__ p, const float* __restrict__ w,
             const float* __restrict__ bias, float* __restrict__ y1) {
    int b = blockIdx.x / 3, o = blockIdx.x % 3;
    for (int e = threadIdx.x; e < Ez; e += 256) {
        float acc = bias[o];
        for (int c = 0; c < Tz; ++c) {
            const float* xr = p + ((long)b * 2 * Tz + c) * Ez;
            const float* wr = w + (o * Tz + c) * 3;
            if (e > 0)      acc += xr[e - 1] * wr[0];
            acc += xr[e] * wr[1];
            if (e < Ez - 1) acc += xr[e + 1] * wr[2];
        }
        y1[(b * 3 + o) * Ez + e] = geluf(acc);
    }
}

// conv2 on transposed (B,E,3) -> h2 (B,NF,3)
__global__ void k_conv2(const float* __restrict__ y1, const float* __restrict__ w,
                        const float* __restrict__ bias, float* __restrict__ h2) {
    int idx = blockIdx.x * blockDim.x + threadIdx.x;
    if (idx >= Bz * NFz * 3) return;
    int pp = idx % 3, o = (idx / 3) % NFz, b = idx / (3 * NFz);
    float acc = bias[o];
    for (int c = 0; c < Ez; ++c) {
        const float* wr = w + (o * Ez + c) * 3;
#pragma unroll
        for (int j = 0; j < 3; ++j) {
            int p2 = pp + j - 1;
            if (p2 >= 0 && p2 < 3) acc += y1[(b * 3 + p2) * Ez + c] * wr[j];
        }
    }
    h2[idx] = acc;
}

// trend[b,t,n] = sum_p h2[b,n,p] * lin^(p+1)
__global__ void k_trend(const float* __restrict__ h2, float* __restrict__ out) {
    int idx = blockIdx.x * blockDim.x + threadIdx.x;  // B*T*NF
    int n = idx % NFz, t = (idx / NFz) % Tz, b = idx / (NFz * Tz);
    float lin = (float)(t + 1) / (float)(Tz + 1);
    const float* hr = h2 + (b * NFz + n) * 3;
    out[idx] = hr[0] * lin + hr[1] * lin * lin + hr[2] * lin * lin * lin;
}

// per-(b,d) DFT, top-5 by magnitude, reconstruct season
__global__ __launch_bounds__(256)
void k_fourier(const float* __restrict__ p, float* __restrict__ season) {
    __shared__ float xr[512], ctab[512], stab[512];
    __shared__ float mg[256], re[256], im[256];
    __shared__ float rv[256]; __shared__ int ri[256];
    __shared__ float selC[5], selS[5]; __shared__ int selF[5];
    int d = blockIdx.x & (Ez - 1), b = blockIdx.x >> 10;
    int tid = threadIdx.x;
    const float* x2 = p + ((long)(b * 2 * Tz + Tz)) * Ez + d;
#pragma unroll
    for (int i = 0; i < 2; ++i) {
        int t = tid + i * 256;
        xr[t] = x2[(long)t * Ez];
        float ang = 6.2831853071795864f * (float)t / 512.0f;
        ctab[t] = cosf(ang); stab[t] = sinf(ang);
    }
    __syncthreads();
    float fre = 0.f, fim = 0.f;
    int f = tid;
    bool valid = (f >= 1) && (f <= 255);
    if (valid) {
        for (int t = 0; t < 512; ++t) {
            int j = (f * t) & 511;
            fre += xr[t] * ctab[j];
            fim -= xr[t] * stab[j];
        }
    }
    mg[tid] = valid ? sqrtf(fre * fre + fim * fim) : -1.0f;
    re[tid] = fre; im[tid] = fim;
    __syncthreads();
    for (int sel = 0; sel < 5; ++sel) {
        rv[tid] = mg[tid]; ri[tid] = tid; __syncthreads();
        for (int st = 128; st > 0; st >>= 1) {
            if (tid < st && rv[tid + st] > rv[tid]) { rv[tid] = rv[tid + st]; ri[tid] = ri[tid + st]; }
            __syncthreads();
        }
        if (tid == 0) {
            int fi = ri[0];
            selF[sel] = fi;
            selC[sel] = re[fi];   // amp*cos(phase)
            selS[sel] = im[fi];   // amp*sin(phase)
            mg[fi] = -2.0f;
        }
        __syncthreads();
    }
#pragma unroll
    for (int i = 0; i < 2; ++i) {
        int t = tid + i * 256;
        float acc = 0.f;
#pragma unroll
        for (int s = 0; s < 5; ++s) {
            int j = (selF[s] * t) & 511;
            acc += selC[s] * ctab[j] - selS[s] * stab[j];
        }
        season[((long)(b * Tz) + t) * Ez + d] = 2.0f * acc;   // + conjugate pair
    }
}

__global__ void k_mean(const float* __restrict__ x, float* __restrict__ m) {
    int b = blockIdx.x;
    for (int e = threadIdx.x; e < Ez; e += blockDim.x) {
        float s = 0.f;
        const float* xb = x + (long)b * Tz * Ez + e;
        for (int t = 0; t < Tz; ++t) s += xb[(long)t * Ez];
        m[b * Ez + e] = s * (1.0f / Tz);
    }
}

__global__ void k_sub(const float* __restrict__ x, const float* __restrict__ m,
                      float* __restrict__ out) {
    long idx = (long)blockIdx.x * blockDim.x + threadIdx.x;
    int e = (int)(idx & (Ez - 1));
    int b = (int)(idx / ((long)Tz * Ez));
    out[idx] = x[idx] - m[b * Ez + e];
}

// ---------------------------------------------------------------------------
extern "C" void kernel_launch(void* const* d_in, const int* in_sizes, int n_in,
                              void* d_out, int out_size, void* d_ws, size_t ws_size,
                              hipStream_t stream) {
    (void)in_sizes; (void)n_in; (void)out_size; (void)ws_size;

    const float* x_in   = (const float*)d_in[0];
    const float* enc    = (const float*)d_in[1];
    const int*   ts     = (const int*)d_in[2];
    const float* ln1_w  = (const float*)d_in[3];  const float* ln1_b  = (const float*)d_in[4];
    const float* ln11_w = (const float*)d_in[5];  const float* ln11_b = (const float*)d_in[6];
    const float* ln2_w  = (const float*)d_in[7];  const float* ln2_b  = (const float*)d_in[8];
    const float* a1_qkv_w = (const float*)d_in[9];  const float* a1_qkv_b = (const float*)d_in[10];
    const float* a1_agq_w = (const float*)d_in[11]; const float* a1_agq_b = (const float*)d_in[12];
    const float* a1_agk_w = (const float*)d_in[13]; const float* a1_agk_b = (const float*)d_in[14];
    const float* a1_agv_w = (const float*)d_in[15]; const float* a1_agv_b = (const float*)d_in[16];
    const float* a1_ago_w = (const float*)d_in[17]; const float* a1_ago_b = (const float*)d_in[18];
    const float* a1_agents = (const float*)d_in[19];
    const float* a1_proj_w = (const float*)d_in[20]; const float* a1_proj_b = (const float*)d_in[21];
    const float* a2_q_w  = (const float*)d_in[22]; const float* a2_q_b  = (const float*)d_in[23];
    const float* a2_kv_w = (const float*)d_in[24]; const float* a2_kv_b = (const float*)d_in[25];
    const float* a2_agq_w = (const float*)d_in[26]; const float* a2_agq_b = (const float*)d_in[27];
    const float* a2_agk_w = (const float*)d_in[28]; const float* a2_agk_b = (const float*)d_in[29];
    const float* a2_agv_w = (const float*)d_in[30]; const float* a2_agv_b = (const float*)d_in[31];
    const float* a2_ago_w = (const float*)d_in[32]; const float* a2_ago_b = (const float*)d_in[33];
    const float* a2_agents = (const float*)d_in[34];
    const float* a2_proj_w = (const float*)d_in[35]; const float* a2_proj_b = (const float*)d_in[36];
    const float* tr_c1_w = (const float*)d_in[37]; const float* tr_c1_b = (const float*)d_in[38];
    const float* tr_c2_w = (const float*)d_in[39]; const float* tr_c2_b = (const float*)d_in[40];
    const float* mlp_w1 = (const float*)d_in[41]; const float* mlp_b1 = (const float*)d_in[42];
    const float* mlp_w2 = (const float*)d_in[43]; const float* mlp_b2 = (const float*)d_in[44];
    const float* pr_w = (const float*)d_in[45]; const float* pr_b = (const float*)d_in[46];
    const float* lin_w = (const float*)d_in[47]; const float* lin_b = (const float*)d_in[48];

    // workspace layout (floats); mlp hidden aliases the dead qkv+s1 region
    float* wsf  = (float*)d_ws;
    float* qkv  = wsf;                       // 3*BTE   (stage2: qc | kv)
    float* s1b  = wsf + 3 * BTE;             // BTE
    float* nb   = wsf + 4 * BTE;             // BTE
    float* kab  = wsf + 5 * BTE;             // BTE
    float* vab  = wsf + 6 * BTE;             // BTE
    float* yb   = wsf + 7 * BTE;             // BTE
    float* xc   = wsf + 8 * BTE;             // BTE
    float* pb   = wsf + 9 * BTE;             // 2*BTE
    float* silu = wsf + 11 * BTE;            // 8192
    float* hln1  = silu + 8192;              // 24576 each
    float* hln11 = hln1 + 24576;
    float* hln2  = hln11 + 24576;
    float* qab = hln2 + 24576;               // 65536
    float* c1b = qab + 65536;                // 524288
    float* y1b = c1b + 524288;               // 24576
    float* h2b = y1b + 24576;                // 1536
    float* mb  = h2b + 1536;                 // 8192
    float* mh  = qkv;                        // alias: 4*BTE mlp hidden

    float* outf = (float*)d_out;
    float* out1 = outf;                      // x - m   (B,T,E)
    float* out2 = outf + BTE;                // m@lin_w (B,1,NF)
    float* out3 = out2 + Bz * NFz;           // trend   (B,T,NF)
    float* out4 = out3 + (long)Bz * Tz * NFz;// season  (B,T,E)

    const int MT = Bz * Tz;                  // 4096

    // time embedding + adaLN modulation vectors
    k_temb<<<Bz, 512, 0, stream>>>(ts, silu);
    k_smallmm<<<(Bz * 3 * Ez + 255) / 256, 256, 0, stream>>>(silu, ln1_w,  ln1_b,  hln1,  Bz, Ez, 3 * Ez);
    k_smallmm<<<(Bz * 3 * Ez + 255) / 256, 256, 0, stream>>>(silu, ln11_w, ln11_b, hln11, Bz, Ez, 3 * Ez);
    k_smallmm<<<(Bz * 3 * Ez + 255) / 256, 256, 0, stream>>>(silu, ln2_w,  ln2_b,  hln2,  Bz, Ez, 3 * Ez);

    // ---- self agent attention
    k_adaln<<<MT, 256, 0, stream>>>(x_in, hln1, nb);
    { dim3 g(3 * Ez / 64, MT / 64, 1);
      k_gemm<0><<<g, 128, 0, stream>>>(nb, a1_qkv_w, a1_qkv_b, 0, nullptr, nullptr, 0, 1,
                                       qkv, MT, 3 * Ez, Ez, 0, 0, 0); }
    k_qa<<<Hz * Az, 64, 0, stream>>>(a1_agents, a1_agq_w, a1_agq_b, qab);
    k_headproj<<<Bz * Hz * Tz, 64, 0, stream>>>(qkv, 3 * Ez, Ez,     a1_agk_w, a1_agk_b, kab);
    k_headproj<<<Bz * Hz * Tz, 64, 0, stream>>>(qkv, 3 * Ez, 2 * Ez, a1_agv_w, a1_agv_b, vab);
    k_s1<<<Bz * Hz * Az, 256, 0, stream>>>(qab, kab, s1b);
    k_c1<<<Bz * Hz * Az, 64, 0, stream>>>(s1b, vab, c1b);
    k_attn2<<<Bz * Hz * Tz, 64, 0, stream>>>(qkv, 3 * Ez, 0, c1b, yb);
    { dim3 g(Ez / 64, MT / 64, 1);
      k_gemm<0><<<g, 128, 0, stream>>>(yb, a1_ago_w, a1_ago_b, 0, nullptr, nullptr, 0, 1,
                                       nb, MT, Ez, Ez, 0, 0, 0);
      k_gemm<2><<<g, 128, 0, stream>>>(nb, a1_proj_w, a1_proj_b, 0, x_in, hln1 + 2 * Ez, 3 * Ez, Tz,
                                       xc, MT, Ez, Ez, 0, 0, 0); }

    // ---- cross agent attention
    k_adaln<<<MT, 256, 0, stream>>>(xc, hln11, nb);
    { dim3 g(Ez / 64, MT / 64, 1);
      k_gemm<0><<<g, 128, 0, stream>>>(nb, a2_q_w, a2_q_b, 0, nullptr, nullptr, 0, 1,
                                       qkv, MT, Ez, Ez, 0, 0, 0); }
    { dim3 g(2 * Ez / 64, MT / 64, 1);
      k_gemm<0><<<g, 128, 0, stream>>>(enc, a2_kv_w, a2_kv_b, 0, nullptr, nullptr, 0, 1,
                                       qkv + BTE, MT, 2 * Ez, Ez, 0, 0, 0); }
    k_qa<<<Hz * Az, 64, 0, stream>>>(a2_agents, a2_agq_w, a2_agq_b, qab);
    k_headproj<<<Bz * Hz * Tz, 64, 0, stream>>>(qkv + BTE, 2 * Ez, 0,  a2_agk_w, a2_agk_b, kab);
    k_headproj<<<Bz * Hz * Tz, 64, 0, stream>>>(qkv + BTE, 2 * Ez, Ez, a2_agv_w, a2_agv_b, vab);
    k_s1<<<Bz * Hz * Az, 256, 0, stream>>>(qab, kab, s1b);
    k_c1<<<Bz * Hz * Az, 64, 0, stream>>>(s1b, vab, c1b);
    k_attn2<<<Bz * Hz * Tz, 64, 0, stream>>>(qkv, Ez, 0, c1b, yb);
    { dim3 g(Ez / 64, MT / 64, 1);
      k_gemm<0><<<g, 128, 0, stream>>>(yb, a2_ago_w, a2_ago_b, 0, nullptr, nullptr, 0, 1,
                                       nb, MT, Ez, Ez, 0, 0, 0);
      k_gemm<2><<<g, 128, 0, stream>>>(nb, a2_proj_w, a2_proj_b, 0, xc, hln11 + 2 * Ez, 3 * Ez, Tz,
                                       xc, MT, Ez, Ez, 0, 0, 0); }

    // ---- trend / season branch: p = pr_w @ x[b]  (batched, row bias)
    { dim3 g(Ez / 64, (2 * Tz) / 64, Bz);
      k_gemm<0><<<g, 128, 0, stream>>>(pr_w, xc, pr_b, 1, nullptr, nullptr, 0, 1,
                                       pb, 2 * Tz, Ez, Tz, 0, (long)Tz * Ez, (long)2 * Tz * Ez); }
    k_conv1<<<Bz * 3, 256, 0, stream>>>(pb, tr_c1_w, tr_c1_b, y1b);
    k_conv2<<<(Bz * NFz * 3 + 63) / 64, 64, 0, stream>>>(y1b, tr_c2_w, tr_c2_b, h2b);
    k_trend<<<(Bz * Tz * NFz) / 256, 256, 0, stream>>>(h2b, out3);
    k_fourier<<<Bz * Ez, 256, 0, stream>>>(pb, out4);

    // ---- MLP
    k_adaln<<<MT, 256, 0, stream>>>(xc, hln2, nb);
    { dim3 g(MLPz / 64, MT / 64, 1);
      k_gemm<1><<<g, 128, 0, stream>>>(nb, mlp_w1, mlp_b1, 0, nullptr, nullptr, 0, 1,
                                       mh, MT, MLPz, Ez, 0, 0, 0); }
    { dim3 g(Ez / 64, MT / 64, 1);
      k_gemm<2><<<g, 128, 0, stream>>>(mh, mlp_w2, mlp_b2, 0, xc, hln2 + 2 * Ez, 3 * Ez, Tz,
                                       xc, MT, Ez, MLPz, 0, 0, 0); }

    // ---- final outputs
    k_mean<<<Bz, 256, 0, stream>>>(xc, mb);
    k_sub<<<(int)(BTE / 256), 256, 0, stream>>>(xc, mb, out1);
    k_smallmm<<<(Bz * NFz + 255) / 256, 256, 0, stream>>>(mb, lin_w, lin_b, out2, Bz, Ez, NFz);
}
